// HybridModel_6485400617742
// MI455X (gfx1250) — matrix operand order repaired and checked by reading, not compiled, not run on hardware
//
#include <hip/hip_runtime.h>
#include <math.h>

typedef __attribute__((ext_vector_type(16))) __bf16 v16bf;
typedef __attribute__((ext_vector_type(8)))  float  v8f;

union BFrag { v16bf v; uint4 q[2]; };

__device__ __forceinline__ unsigned short f2bf(float x) {
  union { float f; unsigned int u; } c; c.f = x;
  unsigned int u = c.u;
  u = u + 0x7FFFu + ((u >> 16) & 1u);   // round-to-nearest-even
  return (unsigned short)(u >> 16);
}

#define BN_EPS 1e-5f

// ---------------------------------------------------------------------------
// Prep: BN-affine folding (conv bias folded in), head constants, bf16 weight
// packing with K-order = (tap, channel) so im2col rows are channel-contiguous.
// ---------------------------------------------------------------------------
__global__ void prep_kernel(
    const float* __restrict__ conv1_b,
    const float* __restrict__ conv2_w, const float* __restrict__ conv2_b,
    const float* __restrict__ conv3_w, const float* __restrict__ conv3_b,
    const float* __restrict__ bn1_g, const float* __restrict__ bn1_b,
    const float* __restrict__ bn1_m, const float* __restrict__ bn1_v,
    const float* __restrict__ bn2_g, const float* __restrict__ bn2_b,
    const float* __restrict__ bn2_m, const float* __restrict__ bn2_v,
    const float* __restrict__ bn3_g, const float* __restrict__ bn3_b,
    const float* __restrict__ bn3_m, const float* __restrict__ bn3_v,
    const float* __restrict__ fc2_w, const float* __restrict__ fc2_b,
    const float* __restrict__ fc3_w, const float* __restrict__ fc3_b,
    float* __restrict__ affine1, float* __restrict__ affine2,
    float* __restrict__ affine3, float* __restrict__ headAB,
    unsigned short* __restrict__ w2p, unsigned short* __restrict__ w3p)
{
  const int tid = threadIdx.x;
  const int gid = blockIdx.x * blockDim.x + tid;
  const int gsz = gridDim.x * blockDim.x;

  if (blockIdx.x == 0) {
    if (tid < 64) {
      float a = bn1_g[tid] * rsqrtf(bn1_v[tid] + BN_EPS);
      affine1[tid]      = a;
      affine1[64 + tid] = (conv1_b[tid] - bn1_m[tid]) * a + bn1_b[tid];
    }
    if (tid < 128) {
      float a = bn2_g[tid] * rsqrtf(bn2_v[tid] + BN_EPS);
      affine2[tid]       = a;
      affine2[128 + tid] = (conv2_b[tid] - bn2_m[tid]) * a + bn2_b[tid];
    }
    if (tid < 256) {
      float a = bn3_g[tid] * rsqrtf(bn3_v[tid] + BN_EPS);
      affine3[tid]       = a;
      affine3[256 + tid] = (conv3_b[tid] - bn3_m[tid]) * a + bn3_b[tid];
    }
    if (tid < 2) {
      // logits[b,c] = q[b]*A[c] + B[c]
      float A = 0.f, Bc = 0.f;
      for (int j = 0; j < 128; ++j) {
        A  += fc3_w[tid * 128 + j] * fc2_w[j];
        Bc += fc3_w[tid * 128 + j] * fc2_b[j];
      }
      headAB[tid]     = A;
      headAB[2 + tid] = Bc + fc3_b[tid];
    }
  }
  // conv2 weights: [128][64][3][3] f32 -> bf16 [o][tap*64 + c]
  for (int i = gid; i < 128 * 576; i += gsz) {
    int o = i / 576, kk = i % 576;
    int tap = kk >> 6, c = kk & 63;
    w2p[i] = f2bf(conv2_w[(o * 64 + c) * 9 + tap]);
  }
  // conv3 weights: [256][128][3][3] f32 -> bf16 [o][tap*128 + c]
  for (int i = gid; i < 256 * 1152; i += gsz) {
    int o = i / 1152, kk = i % 1152;
    int tap = kk >> 7, c = kk & 127;
    w3p[i] = f2bf(conv3_w[(o * 128 + c) * 9 + tap]);
  }
}

// ---------------------------------------------------------------------------
// Stage 1: conv1(1->64, 3x3 SAME) + BN + ReLU + 2x2 maxpool, direct VALU.
// One workgroup per image; output NHWC bf16 [B][14][14][64].
// ---------------------------------------------------------------------------
__global__ void conv1_kernel(const float* __restrict__ x,
                             const float* __restrict__ w1,
                             const float* __restrict__ affine1,
                             unsigned short* __restrict__ act1)
{
  __shared__ float simg[30 * 30];   // zero-padded 28x28
  __shared__ float sw[64 * 9];
  __shared__ float sal[64], sbe[64];
  const int b = blockIdx.x, tid = threadIdx.x;

  for (int i = tid; i < 900; i += 256) simg[i] = 0.f;
  __syncthreads();
  for (int i = tid; i < 784; i += 256) {
    int y = i / 28, xx = i % 28;
    simg[(y + 1) * 30 + (xx + 1)] = x[(size_t)b * 784 + i];
  }
  for (int i = tid; i < 576; i += 256) sw[i] = w1[i];
  if (tid < 64) { sal[tid] = affine1[tid]; sbe[tid] = affine1[64 + tid]; }
  __syncthreads();

  for (int idx = tid; idx < 196 * 64; idx += 256) {
    int p = idx >> 6, c = idx & 63;
    int py = p / 14, px = p % 14;
    float al = sal[c], be = sbe[c];
    const float* wc = &sw[c * 9];
    float best = 0.f;
#pragma unroll
    for (int i2 = 0; i2 < 2; ++i2)
#pragma unroll
      for (int j2 = 0; j2 < 2; ++j2) {
        int y0 = 2 * py + i2, x0 = 2 * px + j2;   // padded coords: (+1 pad, -1 tap) cancel
        float s = 0.f;
#pragma unroll
        for (int ty = 0; ty < 3; ++ty)
#pragma unroll
          for (int tx = 0; tx < 3; ++tx)
            s += simg[(y0 + ty) * 30 + (x0 + tx)] * wc[ty * 3 + tx];
        best = fmaxf(best, fmaxf(s * al + be, 0.f));
      }
    act1[((size_t)b * 196 + p) * 64 + c] = f2bf(best);
  }
}

// ---------------------------------------------------------------------------
// Stage 2: conv2(64->128) implicit GEMM on v_wmma_f32_16x16x32_bf16.
// One workgroup per image. 8 waves = (2 M-halves) x (4 N-pairs): each wave
// owns 7 M-tiles x 2 N-tiles so every A fragment feeds TWO wmmas.
// A fragments double-buffered across the mt loop, B fragments double-buffered
// across K-steps (global loads for ks+1 issued before the wmmas of ks).
// Output pooled NHWC bf16 [B][7][7][128].
// ---------------------------------------------------------------------------
__global__ void __launch_bounds__(256) conv2_kernel(
    const unsigned short* __restrict__ act1,
    const unsigned short* __restrict__ w2p,
    const float* __restrict__ affine2,
    unsigned short* __restrict__ act2)
{
  __shared__ unsigned short halo[16 * 16 * 64];   // 14x14 + zero halo, 32 KB
  const int b = blockIdx.x, tid = threadIdx.x;

  for (int i = tid; i < 256 * 8; i += 256) {      // 256 pixels * 8 uint4
    int pix = i >> 3, qi = i & 7;
    int hy = pix >> 4, hx = pix & 15;
    uint4 val; val.x = val.y = val.z = val.w = 0u;
    int y = hy - 1, xx = hx - 1;
    if ((unsigned)y < 14u && (unsigned)xx < 14u)
      val = *(const uint4*)(act1 + ((size_t)b * 196 + y * 14 + xx) * 64 + qi * 8);
    *(uint4*)(&halo[pix * 64 + qi * 8]) = val;
  }
  __syncthreads();

  const int wave = tid >> 5, lane = tid & 31;
  const int g = lane >> 4, r = lane & 15;
  const int half = wave >> 2;          // M-half: mt 0..6 or 7..13 (13 = dummy)
  const int mtbase = half * 7;
  const int ch0 = (wave & 3) * 32 + r; // two N-tiles: ch0, ch0+16
  const int ch1 = ch0 + 16;
  const float al0 = affine2[ch0], be0 = affine2[128 + ch0];
  const float al1 = affine2[ch1], be1 = affine2[128 + ch1];

  // A-row -> input pixel base (tap-independent); ISA 16-bit A layout: M = lane&15
  int rowoff[7];
#pragma unroll
  for (int i = 0; i < 7; ++i) {
    int m = (mtbase + i) * 16 + r;
    int grp = m >> 2; grp = grp > 48 ? 48 : grp;  // clamp dead rows to safe addrs
    int mem = m & 3;
    int py = grp / 7, px = grp % 7;
    rowoff[i] = ((2 * py + (mem >> 1)) * 16 + (2 * px + (mem & 1))) * 64;
  }

  const v8f vzero = {0.f, 0.f, 0.f, 0.f, 0.f, 0.f, 0.f, 0.f};
  v8f acc0[7], acc1[7];
#pragma unroll
  for (int i = 0; i < 7; ++i) { acc0[i] = vzero; acc1[i] = vzero; }

  const unsigned short* wb0 = w2p + (size_t)ch0 * 576 + g * 16;
  const unsigned short* wb1 = w2p + (size_t)ch1 * 576 + g * 16;

  BFrag B0a, B1a, B0b, B1b;
  B0a.q[0] = ((const uint4*)wb0)[0]; B0a.q[1] = ((const uint4*)wb0)[1];
  B1a.q[0] = ((const uint4*)wb1)[0]; B1a.q[1] = ((const uint4*)wb1)[1];

  auto kstep = [&](int ks, BFrag& Bc0, BFrag& Bc1, BFrag& Bn0, BFrag& Bn1) {
    if (ks + 1 < 18) {                 // prefetch next K-step's B fragments
      const uint4* p0 = (const uint4*)(wb0 + (ks + 1) * 32);
      const uint4* p1 = (const uint4*)(wb1 + (ks + 1) * 32);
      Bn0.q[0] = p0[0]; Bn0.q[1] = p0[1];
      Bn1.q[0] = p1[0]; Bn1.q[1] = p1[1];
    }
    int tap = ks >> 1, c0 = (ks & 1) << 5;
    int ty = tap / 3, tx = tap % 3;
    int tapoff = (ty * 16 + tx) * 64 + c0 + g * 16;
    BFrag Af[2];
    {
      int a0 = rowoff[0] + tapoff;
      Af[0].q[0] = *(const uint4*)(&halo[a0]);
      Af[0].q[1] = *(const uint4*)(&halo[a0 + 8]);
    }
#pragma unroll
    for (int i = 0; i < 7; ++i) {
      if (i + 1 < 7) {                 // double-buffered A prefetch
        int a1 = rowoff[i + 1] + tapoff;
        Af[(i + 1) & 1].q[0] = *(const uint4*)(&halo[a1]);
        Af[(i + 1) & 1].q[1] = *(const uint4*)(&halo[a1 + 8]);
      }
      acc0[i] = __builtin_amdgcn_wmma_f32_16x16x32_bf16(
          false, Af[i & 1].v, false, Bc0.v, (short)0, acc0[i], false, false);
      acc1[i] = __builtin_amdgcn_wmma_f32_16x16x32_bf16(
          false, Af[i & 1].v, false, Bc1.v, (short)0, acc1[i], false, false);
    }
  };

  for (int ks2 = 0; ks2 < 9; ++ks2) {  // 18 K-steps, B buffers ping-pong
    kstep(2 * ks2,     B0a, B1a, B0b, B1b);
    kstep(2 * ks2 + 1, B0b, B1b, B0a, B1a);
  }

  // BN + ReLU + 2x2 maxpool in registers (each lane holds 2 full pool groups)
#pragma unroll
  for (int i = 0; i < 7; ++i) {
    int mt = mtbase + i;
    if (mt < 13) {
      int g0 = 4 * mt + 2 * g;
      float p00 = 0.f, p01 = 0.f, p10 = 0.f, p11 = 0.f;
#pragma unroll
      for (int v = 0; v < 4; ++v) {
        p00 = fmaxf(p00, fmaxf(acc0[i][v] * al0 + be0, 0.f));
        p10 = fmaxf(p10, fmaxf(acc1[i][v] * al1 + be1, 0.f));
      }
#pragma unroll
      for (int v = 4; v < 8; ++v) {
        p01 = fmaxf(p01, fmaxf(acc0[i][v] * al0 + be0, 0.f));
        p11 = fmaxf(p11, fmaxf(acc1[i][v] * al1 + be1, 0.f));
      }
      size_t base = (size_t)b * 49;
      if (g0 <= 48) {
        act2[(base + g0) * 128 + ch0] = f2bf(p00);
        act2[(base + g0) * 128 + ch1] = f2bf(p10);
      }
      if (g0 + 1 <= 48) {
        act2[(base + g0 + 1) * 128 + ch0] = f2bf(p01);
        act2[(base + g0 + 1) * 128 + ch1] = f2bf(p11);
      }
    }
  }
}

// ---------------------------------------------------------------------------
// Stage 3: conv3(128->256) WMMA + BN + ReLU + pool, fused with the single live
// fc1 row (only feature 1 feeds the quantum block). Only the 36/49 pixels that
// survive pooling are computed (M = 3 tiles). Each wave owns 2 N-tiles
// concurrently; A and B fragments double-buffered as in conv2. Emits h1[b].
// ---------------------------------------------------------------------------
__global__ void __launch_bounds__(256) conv3_kernel(
    const unsigned short* __restrict__ act2,
    const unsigned short* __restrict__ w3p,
    const float* __restrict__ affine3,
    const float* __restrict__ fc1_w,
    const float* __restrict__ fc1_b,
    float* __restrict__ h1buf)
{
  __shared__ unsigned short halo[81 * 128];   // 7x7 + zero halo, 20.25 KB
  __shared__ float red[256];
  const int b = blockIdx.x, tid = threadIdx.x;

  for (int i = tid; i < 81 * 16; i += 256) {
    int pix = i >> 4, qi = i & 15;
    int hy = pix / 9, hx = pix % 9;
    uint4 val; val.x = val.y = val.z = val.w = 0u;
    int y = hy - 1, xx = hx - 1;
    if ((unsigned)y < 7u && (unsigned)xx < 7u)
      val = *(const uint4*)(act2 + ((size_t)b * 49 + y * 7 + xx) * 128 + qi * 8);
    *(uint4*)(&halo[pix * 128 + qi * 8]) = val;
  }
  __syncthreads();

  const int wave = tid >> 5, lane = tid & 31;
  const int g = lane >> 4, r = lane & 15;
  const int ch0 = wave * 32 + r;       // two N-tiles: ch0, ch0+16
  const int ch1 = ch0 + 16;
  const float al0 = affine3[ch0], be0 = affine3[256 + ch0];
  const float al1 = affine3[ch1], be1 = affine3[256 + ch1];

  int rowoff[3];
#pragma unroll
  for (int i = 0; i < 3; ++i) {
    int m = i * 16 + r;
    int grp = m >> 2; grp = grp > 8 ? 8 : grp;
    int mem = m & 3;
    int py = grp / 3, px = grp % 3;
    rowoff[i] = ((2 * py + (mem >> 1)) * 9 + (2 * px + (mem & 1))) * 128;
  }

  const v8f vzero = {0.f, 0.f, 0.f, 0.f, 0.f, 0.f, 0.f, 0.f};
  v8f acc0[3], acc1[3];
#pragma unroll
  for (int i = 0; i < 3; ++i) { acc0[i] = vzero; acc1[i] = vzero; }

  const unsigned short* wb0 = w3p + (size_t)ch0 * 1152 + g * 16;
  const unsigned short* wb1 = w3p + (size_t)ch1 * 1152 + g * 16;

  BFrag B0a, B1a, B0b, B1b;
  B0a.q[0] = ((const uint4*)wb0)[0]; B0a.q[1] = ((const uint4*)wb0)[1];
  B1a.q[0] = ((const uint4*)wb1)[0]; B1a.q[1] = ((const uint4*)wb1)[1];

  auto kstep = [&](int ks, BFrag& Bc0, BFrag& Bc1, BFrag& Bn0, BFrag& Bn1) {
    if (ks + 1 < 36) {
      const uint4* p0 = (const uint4*)(wb0 + (ks + 1) * 32);
      const uint4* p1 = (const uint4*)(wb1 + (ks + 1) * 32);
      Bn0.q[0] = p0[0]; Bn0.q[1] = p0[1];
      Bn1.q[0] = p1[0]; Bn1.q[1] = p1[1];
    }
    int tap = ks >> 2, c0 = (ks & 3) << 5;
    int ty = tap / 3, tx = tap % 3;
    int tapoff = (ty * 9 + tx) * 128 + c0 + g * 16;
    BFrag Af[2];
    {
      int a0 = rowoff[0] + tapoff;
      Af[0].q[0] = *(const uint4*)(&halo[a0]);
      Af[0].q[1] = *(const uint4*)(&halo[a0 + 8]);
    }
#pragma unroll
    for (int i = 0; i < 3; ++i) {
      if (i + 1 < 3) {
        int a1 = rowoff[i + 1] + tapoff;
        Af[(i + 1) & 1].q[0] = *(const uint4*)(&halo[a1]);
        Af[(i + 1) & 1].q[1] = *(const uint4*)(&halo[a1 + 8]);
      }
      acc0[i] = __builtin_amdgcn_wmma_f32_16x16x32_bf16(
          false, Af[i & 1].v, false, Bc0.v, (short)0, acc0[i], false, false);
      acc1[i] = __builtin_amdgcn_wmma_f32_16x16x32_bf16(
          false, Af[i & 1].v, false, Bc1.v, (short)0, acc1[i], false, false);
    }
  };

  for (int ks2 = 0; ks2 < 18; ++ks2) {  // 36 K-steps (9 taps * 128 ch)
    kstep(2 * ks2,     B0a, B1a, B0b, B1b);
    kstep(2 * ks2 + 1, B0b, B1b, B0a, B1a);
  }

  const float* w1row = fc1_w + 2304;   // fc1 row 1 (the only live output)
  float part = 0.f;
#pragma unroll
  for (int i = 0; i < 3; ++i) {
    int g0 = 4 * i + 2 * g;
    float p00 = 0.f, p01 = 0.f, p10 = 0.f, p11 = 0.f;
#pragma unroll
    for (int v = 0; v < 4; ++v) {
      p00 = fmaxf(p00, fmaxf(acc0[i][v] * al0 + be0, 0.f));
      p10 = fmaxf(p10, fmaxf(acc1[i][v] * al1 + be1, 0.f));
    }
#pragma unroll
    for (int v = 4; v < 8; ++v) {
      p01 = fmaxf(p01, fmaxf(acc0[i][v] * al0 + be0, 0.f));
      p11 = fmaxf(p11, fmaxf(acc1[i][v] * al1 + be1, 0.f));
    }
    if (g0 <= 8)
      part += p00 * w1row[ch0 * 9 + g0] + p10 * w1row[ch1 * 9 + g0];
    if (g0 + 1 <= 8)
      part += p01 * w1row[ch0 * 9 + g0 + 1] + p11 * w1row[ch1 * 9 + g0 + 1];
  }

  red[tid] = part;
  __syncthreads();
  for (int s = 128; s > 0; s >>= 1) {
    if (tid < s) red[tid] += red[tid + s];
    __syncthreads();
  }
  if (tid == 0) h1buf[b] = red[0] + fc1_b[1];
}

// ---------------------------------------------------------------------------
// Head: q = cos(h1)*cos(qp[1]); logits = q*A + B; 2-class log_softmax.
// ---------------------------------------------------------------------------
__global__ void head_kernel(const float* __restrict__ h1buf,
                            const float* __restrict__ q_params,
                            const float* __restrict__ headAB,
                            float* __restrict__ out, int B)
{
  int i = blockIdx.x * blockDim.x + threadIdx.x;
  if (i >= B) return;
  float q  = cosf(h1buf[i]) * cosf(q_params[1]);
  float l0 = q * headAB[0] + headAB[2];
  float l1 = q * headAB[1] + headAB[3];
  float m   = fmaxf(l0, l1);
  float lse = m + logf(expf(l0 - m) + expf(l1 - m));
  out[2 * i]     = l0 - lse;
  out[2 * i + 1] = l1 - lse;
}

// ---------------------------------------------------------------------------
extern "C" void kernel_launch(void* const* d_in, const int* in_sizes, int n_in,
                              void* d_out, int out_size, void* d_ws, size_t ws_size,
                              hipStream_t stream)
{
  const float* x        = (const float*)d_in[0];
  const float* conv1_w  = (const float*)d_in[1];
  const float* conv1_b  = (const float*)d_in[2];
  const float* conv2_w  = (const float*)d_in[3];
  const float* conv2_b  = (const float*)d_in[4];
  const float* conv3_w  = (const float*)d_in[5];
  const float* conv3_b  = (const float*)d_in[6];
  const float* bn1_g = (const float*)d_in[7];
  const float* bn1_b = (const float*)d_in[8];
  const float* bn1_m = (const float*)d_in[9];
  const float* bn1_v = (const float*)d_in[10];
  const float* bn2_g = (const float*)d_in[11];
  const float* bn2_b = (const float*)d_in[12];
  const float* bn2_m = (const float*)d_in[13];
  const float* bn2_v = (const float*)d_in[14];
  const float* bn3_g = (const float*)d_in[15];
  const float* bn3_b = (const float*)d_in[16];
  const float* bn3_m = (const float*)d_in[17];
  const float* bn3_v = (const float*)d_in[18];
  const float* fc1_w    = (const float*)d_in[19];
  const float* fc1_b    = (const float*)d_in[20];
  const float* q_params = (const float*)d_in[21];
  const float* fc2_w    = (const float*)d_in[22];
  const float* fc2_b    = (const float*)d_in[23];
  const float* fc3_w    = (const float*)d_in[24];
  const float* fc3_b    = (const float*)d_in[25];
  (void)n_in; (void)out_size; (void)ws_size;

  const int B = in_sizes[0] / 784;   // 8192

  char* ws = (char*)d_ws;
  size_t off = 0;
  float* affine1 = (float*)(ws + off); off += 128 * 4;
  float* affine2 = (float*)(ws + off); off += 256 * 4;
  float* affine3 = (float*)(ws + off); off += 512 * 4;
  float* headAB  = (float*)(ws + off); off += 4 * 4;
  off = (off + 255) & ~(size_t)255;
  unsigned short* w2p = (unsigned short*)(ws + off); off += (size_t)128 * 576 * 2;
  unsigned short* w3p = (unsigned short*)(ws + off); off += (size_t)256 * 1152 * 2;
  off = (off + 255) & ~(size_t)255;
  float* h1buf = (float*)(ws + off); off += (size_t)B * 4;
  off = (off + 255) & ~(size_t)255;
  unsigned short* act1 = (unsigned short*)(ws + off); off += (size_t)B * 196 * 64 * 2;
  unsigned short* act2 = (unsigned short*)(ws + off); off += (size_t)B * 49 * 128 * 2;

  prep_kernel<<<128, 256, 0, stream>>>(
      conv1_b, conv2_w, conv2_b, conv3_w, conv3_b,
      bn1_g, bn1_b, bn1_m, bn1_v,
      bn2_g, bn2_b, bn2_m, bn2_v,
      bn3_g, bn3_b, bn3_m, bn3_v,
      fc2_w, fc2_b, fc3_w, fc3_b,
      affine1, affine2, affine3, headAB, w2p, w3p);

  conv1_kernel<<<B, 256, 0, stream>>>(x, conv1_w, affine1, act1);
  conv2_kernel<<<B, 256, 0, stream>>>(act1, w2p, affine2, act2);
  conv3_kernel<<<B, 256, 0, stream>>>(act2, w3p, affine3, fc1_w, fc1_b, h1buf);
  head_kernel<<<(B + 255) / 256, 256, 0, stream>>>(h1buf, q_params, headAB,
                                                   (float*)d_out, B);
}
